// QuatE_49649821942232
// MI455X (gfx1250) — matrix operand order repaired and checked by reading, not compiled, not run on hardware
//
#include <hip/hip_runtime.h>
#include <hip/hip_bf16.h>
#include <math.h>

typedef __attribute__((ext_vector_type(16))) __bf16 v16bf;
typedef __attribute__((ext_vector_type(8)))  __bf16 v8bf;
typedef __attribute__((ext_vector_type(2)))  __bf16 v2bf;
typedef __attribute__((ext_vector_type(8)))  float  v8f;

#define BM 128
#define BN 128
#define BK 32
#define LDA 40   // bf16 elems per LDS row (BK + 8 pad -> 80B stride)

// Pack two f32 into two bf16 (one uint).
__device__ __forceinline__ unsigned int pk2(float a, float b) {
#if __has_builtin(__builtin_amdgcn_cvt_pk_bf16_f32)
  v2bf v = __builtin_amdgcn_cvt_pk_bf16_f32(a, b);
  return __builtin_bit_cast(unsigned int, v);
#else
  unsigned int ua = __builtin_bit_cast(unsigned int, a);
  unsigned int ub = __builtin_bit_cast(unsigned int, b);
  return (ua >> 16) | (ub & 0xFFFF0000u);
#endif
}

// f32 -> bf16 with optional K padding (K -> Kp, zero fill). dst rows have Kp elems.
__global__ __launch_bounds__(256)
void cvt_pad_bf16(const float* __restrict__ src, unsigned int* __restrict__ dst,
                  int K, int Kp, int pairs) {
  int i = blockIdx.x * blockDim.x + threadIdx.x;
  const int stride = gridDim.x * blockDim.x;
  if (K == Kp) {
    for (; i < pairs; i += stride)
      dst[i] = pk2(src[(size_t)2 * i], src[(size_t)2 * i + 1]);
  } else {
    const int kp2 = Kp >> 1;
    for (; i < pairs; i += stride) {
      int row = i / kp2;
      int c0  = (i - row * kp2) * 2;
      const float* sp = src + (size_t)row * K;
      float a = (c0     < K) ? sp[c0]     : 0.0f;
      float b = (c0 + 1 < K) ? sp[c0 + 1] : 0.0f;
      dst[i] = pk2(a, b);
    }
  }
}

// 2x4 WMMA tiles per wave from LDS tiles As (128 x BK) / Bs (128 x BK).
__device__ __forceinline__ void wmma_tiles(const unsigned short* __restrict__ As,
                                           const unsigned short* __restrict__ Bs,
                                           int aRowBase, int bColBase, int fRow,
                                           int aKlo, int bKlo, v8f acc[2][4]) {
  v16bf aF[2], bF[4];
#pragma unroll
  for (int tm = 0; tm < 2; ++tm) {
    const unsigned short* p = As + (aRowBase + tm * 16 + fRow) * LDA + aKlo;
    v8bf lo = *(const v8bf*)p;
    v8bf hi = *(const v8bf*)(p + 16);
    aF[tm] = __builtin_shufflevector(lo, hi, 0,1,2,3,4,5,6,7,8,9,10,11,12,13,14,15);
  }
#pragma unroll
  for (int tn = 0; tn < 4; ++tn) {
    const unsigned short* p = Bs + (bColBase + tn * 16 + fRow) * LDA + bKlo;
    v8bf lo = *(const v8bf*)p;
    v8bf hi = *(const v8bf*)(p + 16);
    bF[tn] = __builtin_shufflevector(lo, hi, 0,1,2,3,4,5,6,7,8,9,10,11,12,13,14,15);
  }
#pragma unroll
  for (int tm = 0; tm < 2; ++tm)
#pragma unroll
    for (int tn = 0; tn < 4; ++tn)
      acc[tm][tn] = __builtin_amdgcn_wmma_f32_16x16x32_bf16(
          false, aF[tm], false, bF[tn], (short)0, acc[tm][tn], false, false);
}

// C[m, blockIdx.x*128 + n] = A[m,:] . Wsel[n0+n,:]  ("NT" GEMM, K-contiguous both sides)
template <bool BF16IN>
__global__ __launch_bounds__(256)
void gemm4(const void* __restrict__ Ap,
           const void* __restrict__ W0, const void* __restrict__ W1,
           const void* __restrict__ W2, const void* __restrict__ W3,
           float* __restrict__ C, int K) {
  __shared__ unsigned short As[2][BM * LDA];
  __shared__ unsigned short Bs[2][BN * LDA];

  const int tid   = threadIdx.x;
  const int lane  = tid & 31;
  const int wv    = tid >> 5;
  const int waveM = wv & 3;    // 4 x 32 rows
  const int waveN = wv >> 2;   // 2 x 64 cols

  const int m0   = blockIdx.y * BM;
  const int wsel = blockIdx.x >> 1;
  const int n0   = (blockIdx.x & 1) * BN;
  const void* W = (wsel == 0) ? W0 : (wsel == 1) ? W1 : (wsel == 2) ? W2 : W3;

  // staging: 256 threads cover 128x32 A and 128x32 B, 16 elems each
  const int arow = tid >> 1;
  const int akof = (tid & 1) * 16;

  const int nk = (K + BK - 1) / BK;

  // fragment addressing (CDNA5 bf16 WMMA per-lane layout)
  const int fRow = lane & 15;
  const int aKlo = (lane < 16) ? 0 : 8;
  const int bKlo = (lane < 16) ? 0 : 8;
  const int aRowBase = waveM * 32;
  const int bColBase = waveN * 64;

  v8f acc[2][4];
#pragma unroll
  for (int a_ = 0; a_ < 2; ++a_)
#pragma unroll
    for (int b_ = 0; b_ < 4; ++b_)
#pragma unroll
      for (int i = 0; i < 8; ++i) acc[a_][b_][i] = 0.0f;

  if constexpr (BF16IN) {
    // ---- pipelined path: bf16 operands, Kp % 32 == 0, no guards ----
    const unsigned short* aG = (const unsigned short*)Ap + (size_t)(m0 + arow) * K + akof;
    const unsigned short* bG = (const unsigned short*)W  + (size_t)(n0 + arow) * K + akof;

    uint4 a0, a1, b0, b1;  // explicit scalars: must stay in VGPRs
    {
      const uint4* qa = (const uint4*)aG;
      a0 = qa[0]; a1 = qa[1];
      const uint4* qb = (const uint4*)bG;
      b0 = qb[0]; b1 = qb[1];
    }
    int buf = 0;
    for (int kt = 0; kt < nk; ++kt) {
      unsigned short* da = &As[buf][arow * LDA + akof];
      unsigned short* db = &Bs[buf][arow * LDA + akof];
      *(uint4*)da = a0; *(uint4*)(da + 8) = a1;
      *(uint4*)db = b0; *(uint4*)(db + 8) = b1;
      __syncthreads();                        // single barrier (double buffered)
      if (kt + 1 < nk) {
        const uint4* qa = (const uint4*)(aG + (kt + 1) * BK);
        a0 = qa[0]; a1 = qa[1];
        const uint4* qb = (const uint4*)(bG + (kt + 1) * BK);
        b0 = qb[0]; b1 = qb[1];
      }
      wmma_tiles(As[buf], Bs[buf], aRowBase, bColBase, fRow, aKlo, bKlo, acc);
      buf ^= 1;
    }
  } else {
    // ---- fallback: f32 operands, convert on the fly, simple 2-barrier loop ----
    const float* aG = (const float*)Ap + (size_t)(m0 + arow) * K + akof;
    const float* bG = (const float*)W  + (size_t)(n0 + arow) * K + akof;
    for (int kt = 0; kt < nk; ++kt) {
      const int k0 = kt * BK;
      unsigned int* da = (unsigned int*)&As[0][arow * LDA + akof];
      unsigned int* db = (unsigned int*)&Bs[0][arow * LDA + akof];
      __syncthreads();
      if (k0 + akof + 16 <= K) {
        const float4* pa = (const float4*)(aG + k0);
        const float4* pb = (const float4*)(bG + k0);
#pragma unroll
        for (int i = 0; i < 4; ++i) {
          float4 va = pa[i], vb = pb[i];
          da[2*i]   = pk2(va.x, va.y); da[2*i+1] = pk2(va.z, va.w);
          db[2*i]   = pk2(vb.x, vb.y); db[2*i+1] = pk2(vb.z, vb.w);
        }
      } else {
#pragma unroll
        for (int i = 0; i < 8; ++i) {
          int k = k0 + akof + 2 * i;
          float a0f = (k     < K) ? aG[k0 + 2*i]     : 0.0f;
          float a1f = (k + 1 < K) ? aG[k0 + 2*i + 1] : 0.0f;
          float b0f = (k     < K) ? bG[k0 + 2*i]     : 0.0f;
          float b1f = (k + 1 < K) ? bG[k0 + 2*i + 1] : 0.0f;
          da[i] = pk2(a0f, a1f);
          db[i] = pk2(b0f, b1f);
        }
      }
      __syncthreads();
      wmma_tiles(As[0], Bs[0], aRowBase, bColBase, fRow, aKlo, bKlo, acc);
    }
  }

  const int cRow0 = m0 + waveM * 32 + ((lane >> 4) << 3);
  const int cCol0 = blockIdx.x * BN + waveN * 64 + (lane & 15);
#pragma unroll
  for (int tm = 0; tm < 2; ++tm)
#pragma unroll
    for (int tn = 0; tn < 4; ++tn)
#pragma unroll
      for (int j = 0; j < 8; ++j)
        C[(size_t)(cRow0 + tm*16 + j) * 1024 + (cCol0 + tn*16)] = acc[tm][tn][j];
}

// One wave per row. score = ||R|| * sum_d (H (x) R) . T ; sigmoid.
__global__ __launch_bounds__(256)
void quate_score(const float* __restrict__ H, const float* __restrict__ T,
                 const float* __restrict__ R, float* __restrict__ out) {
  const int lane = threadIdx.x & 31;
  const int row  = blockIdx.x * 8 + (threadIdx.x >> 5);
  const float* h = H + (size_t)row * 1024;
  const float* t = T + (size_t)row * 1024;
  const float* r = R + (size_t)row * 1024;

  float nrm = 0.0f, s = 0.0f;
#pragma unroll
  for (int d = lane; d < 256; d += 32) {
    float rr = r[d], ri = r[256 + d], rj = r[512 + d], rk = r[768 + d];
    nrm += rr*rr + ri*ri + rj*rj + rk*rk;
    float hr = h[d], hi = h[256 + d], hj = h[512 + d], hk = h[768 + d];
    float qr = hr*rr - hi*ri - hj*rj - hk*rk;
    float qi = hr*ri + hi*rr + hj*rk - hk*rj;
    float qj = hr*rj - hi*rk + hj*rr + hk*ri;
    float qk = hr*rk + hi*rj - hj*ri + hk*rr;
    s += qr*t[d] + qi*t[256 + d] + qj*t[512 + d] + qk*t[768 + d];
  }
#pragma unroll
  for (int off = 16; off > 0; off >>= 1) {
    nrm += __shfl_xor(nrm, off, 32);
    s   += __shfl_xor(s,   off, 32);
  }
  if (lane == 0)
    out[row] = 1.0f / (1.0f + __expf(-(sqrtf(nrm) * s)));
}

extern "C" void kernel_launch(void* const* d_in, const int* in_sizes, int n_in,
                              void* d_out, int out_size, void* d_ws, size_t ws_size,
                              hipStream_t stream) {
  const float* h = (const float*)d_in[0];
  const float* t = (const float*)d_in[1];
  const float* r = (const float*)d_in[2];
  const float* We[4] = {(const float*)d_in[3], (const float*)d_in[4],
                        (const float*)d_in[5], (const float*)d_in[6]};
  const float* Wr[4] = {(const float*)d_in[7], (const float*)d_in[8],
                        (const float*)d_in[9], (const float*)d_in[10]};

  char* w = (char*)d_ws;
  float* Hb = (float*)w;                         // [2048,1024] f32
  float* Tb = Hb + (size_t)2048 * 1024;
  float* Rb = Tb + (size_t)2048 * 1024;
  const size_t f32need = (size_t)3 * 2048 * 1024 * 4;          // 25,165,824

  unsigned short* hbf  = (unsigned short*)(w + f32need);
  unsigned short* tbf  = hbf + (size_t)2048 * 20000;
  unsigned short* webf = tbf + (size_t)2048 * 20000;           // 4 x [256,20000]
  unsigned short* rbf  = webf + (size_t)4 * 256 * 20000;       // [2048,512] padded
  unsigned short* wrbf = rbf + (size_t)2048 * 512;             // 4 x [256,512] padded
  const size_t need = f32need +
      ((size_t)2 * 2048 * 20000 + (size_t)4 * 256 * 20000 +
       (size_t)2048 * 512 + (size_t)4 * 256 * 512) * 2;        // ~233 MB

  dim3 blk(256);
  dim3 grid(8, 16);

  if (ws_size >= need) {
    auto cvt = [&](const float* s, unsigned short* d, int K, int Kp, int rows) {
      int pairs = rows * (Kp >> 1);
      int g = (pairs + 255) / 256; if (g > 8192) g = 8192;
      cvt_pad_bf16<<<g, 256, 0, stream>>>(s, (unsigned int*)d, K, Kp, pairs);
    };
    cvt(h, hbf, 20000, 20000, 2048);
    cvt(t, tbf, 20000, 20000, 2048);
    for (int i = 0; i < 4; ++i) cvt(We[i], webf + (size_t)i * 256 * 20000, 20000, 20000, 256);
    cvt(r, rbf, 500, 512, 2048);
    for (int i = 0; i < 4; ++i) cvt(Wr[i], wrbf + (size_t)i * 256 * 512, 500, 512, 256);

    unsigned short* we0 = webf;
    gemm4<true><<<grid, blk, 0, stream>>>(hbf, we0, we0 + (size_t)1*256*20000,
        we0 + (size_t)2*256*20000, we0 + (size_t)3*256*20000, Hb, 20000);
    gemm4<true><<<grid, blk, 0, stream>>>(tbf, we0, we0 + (size_t)1*256*20000,
        we0 + (size_t)2*256*20000, we0 + (size_t)3*256*20000, Tb, 20000);
    gemm4<true><<<grid, blk, 0, stream>>>(rbf, wrbf, wrbf + (size_t)1*256*512,
        wrbf + (size_t)2*256*512, wrbf + (size_t)3*256*512, Rb, 512);
  } else {
    gemm4<false><<<grid, blk, 0, stream>>>(h, We[0], We[1], We[2], We[3], Hb, 20000);
    gemm4<false><<<grid, blk, 0, stream>>>(t, We[0], We[1], We[2], We[3], Tb, 20000);
    gemm4<false><<<grid, blk, 0, stream>>>(r, Wr[0], Wr[1], Wr[2], Wr[3], Rb, 500);
  }

  quate_score<<<dim3(2048 / 8), blk, 0, stream>>>(Hb, Tb, Rb, (float*)d_out);
}
